// Attention_79740362817946
// MI455X (gfx1250) — matrix operand order repaired and checked by reading, not compiled, not run on hardware
//
#include <hip/hip_runtime.h>
#include <hip/hip_bf16.h>

typedef __attribute__((ext_vector_type(16))) _Float16 v16h;
typedef __attribute__((ext_vector_type(8)))  _Float16 v8h;
typedef __attribute__((ext_vector_type(8)))  float    v8f;
typedef __attribute__((ext_vector_type(4)))  float    v4f;

#define BB 2
#define TT 12
#define RR 10000
#define KK 8
#define DD 64
#define NROWS (BB*TT*RR)      /* 240000 */
#define NTILES (NROWS/16)     /* 15000  */
#define RT_PER_BT (RR/16)     /* 625    */

__device__ __forceinline__ v16h cat8(v8h lo, v8h hi) {
  v16h r;
#pragma unroll
  for (int i = 0; i < 8; ++i) { r[i] = lo[i]; r[i + 8] = hi[i]; }
  return r;
}

__device__ __forceinline__ v8f wmma_f16(v16h a, v16h b, v8f c) {
  // D(f32 16x16) = A(f16 16x32) * B(f16 32x16) + C
  return __builtin_amdgcn_wmma_f32_16x16x32_f16(false, a, false, b, (short)0, c,
                                                false, false);
}

// A fragment (16x32 K-slice) built from f32 global rows, converting to f16.
// 16-bit A layout: lanes 0-15 hold row=lane, K = {s*32+0..7, s*32+16..23};
// lanes 16-31 hold row=lane-16, K = {s*32+8..15, s*32+24..31}.
__device__ __forceinline__ v16h a_frag_from_global_f32(const float* __restrict__ tb,
                                                       int lane, int s) {
  const int row = lane & 15;
  const int hk  = (lane >> 4) << 3;
  const float* rp = tb + (size_t)row * DD + s * 32 + hk;
  v4f a0 = *(const v4f*)(rp +  0);
  v4f a1 = *(const v4f*)(rp +  4);
  v4f a2 = *(const v4f*)(rp + 16);
  v4f a3 = *(const v4f*)(rp + 20);
  v16h A;
#pragma unroll
  for (int i = 0; i < 4; ++i) {
    A[i]      = (_Float16)a0[i];
    A[i + 4]  = (_Float16)a1[i];
    A[i + 8]  = (_Float16)a2[i];
    A[i + 12] = (_Float16)a3[i];
  }
  return A;
}

// A fragment from a 16x64 f16 row-major LDS tile.
__device__ __forceinline__ v16h a_frag_from_lds_f16(const _Float16* Ah, int lane, int s) {
  const int row = lane & 15;
  const int hk  = (lane >> 4) << 3;
  const _Float16* p = Ah + row * DD + s * 32 + hk;
  v8h lo = *(const v8h*)(p);        // K = s*32+hk .. +7
  v8h hi = *(const v8h*)(p + 16);   // K = s*32+hk+16 .. +23
  return cat8(lo, hi);
}

// B fragment from an LDS matrix stored [n][k] (column n holds K contiguous).
// B layout: lanes 0-15: col = n, K = {s*32+0..7, s*32+16..23}; lanes 16-31: col = n, K = {s*32+8..15, s*32+24..31}.
__device__ __forceinline__ v16h b_frag_from_lds(const _Float16* Wt, int lane, int nb, int s) {
  const int c  = nb * 16 + (lane & 15);
  const int k0 = s * 32 + ((lane >> 4) << 3);
  v8h b0 = *(const v8h*)(Wt + c * DD + k0);
  v8h b1 = *(const v8h*)(Wt + c * DD + k0 + 16);
  return cat8(b0, b1);
}

// Async DMA of one 128B row: global -> LDS, tracked by ASYNCcnt.
// INST_OFFSET is added to both the LDS (VDST) and global (VADDR) addresses,
// so one base covers the whole row via offset:0..112.
__device__ __forceinline__ void async_row128_to_lds(const void* gsrc, void* ldst) {
  uint32_t la = (uint32_t)(uintptr_t)ldst;   // flat->LDS truncation = group-seg offset
  asm volatile(
      "global_load_async_to_lds_b128 %0, %1, off\n\t"
      "global_load_async_to_lds_b128 %0, %1, off offset:16\n\t"
      "global_load_async_to_lds_b128 %0, %1, off offset:32\n\t"
      "global_load_async_to_lds_b128 %0, %1, off offset:48\n\t"
      "global_load_async_to_lds_b128 %0, %1, off offset:64\n\t"
      "global_load_async_to_lds_b128 %0, %1, off offset:80\n\t"
      "global_load_async_to_lds_b128 %0, %1, off offset:96\n\t"
      "global_load_async_to_lds_b128 %0, %1, off offset:112"
      :: "v"(la), "v"(gsrc) : "memory");
}

__device__ __forceinline__ void wait_async_done() {
  asm volatile("s_wait_asynccnt 0" ::: "memory");
}

// ---------------------------------------------------------------------------
// Kernel 1: Kp[n][e] = f16( data[n][:] @ wk )  for all 240000 rows.
// One wave per 16-row tile, 8 waves per block.
// ---------------------------------------------------------------------------
__global__ void __launch_bounds__(256) kproj_kernel(const float* __restrict__ data,
                                                    const float* __restrict__ wk,
                                                    _Float16* __restrict__ kp) {
  __shared__ __align__(16) _Float16 wkT[DD * DD];   // wkT[n*64+k] = wk[k][n], f16
  const int tid = threadIdx.x;
#pragma unroll
  for (int i = 0; i < 16; ++i) {
    int id = i * 256 + tid;
    int n = id >> 6, k = id & 63;
    wkT[id] = (_Float16)wk[k * DD + n];
  }
  __syncthreads();

  const int lane = tid & 31;
  const int wave = tid >> 5;
  const size_t base = ((size_t)blockIdx.x * 8 + wave) * 16;
  const float* tb = data + base * DD;

  v8f acc[4] = {};
#pragma unroll
  for (int s = 0; s < 2; ++s) {
    v16h A = a_frag_from_global_f32(tb, lane, s);
    v16h Bf[4];
#pragma unroll
    for (int nb = 0; nb < 4; ++nb) Bf[nb] = b_frag_from_lds(wkT, lane, nb, s);
#pragma unroll
    for (int nb = 0; nb < 4; ++nb) acc[nb] = wmma_f16(A, Bf[nb], acc[nb]);
  }

  // C layout: VGPR j holds M=j (lanes 0-15) / M=j+8 (lanes 16-31), N = lane&15 in block nb
#pragma unroll
  for (int nb = 0; nb < 4; ++nb) {
    const int col = nb * 16 + (lane & 15);
#pragma unroll
    for (int j = 0; j < 8; ++j) {
      const int m = j + ((lane >> 4) << 3);
      kp[(base + m) * DD + col] = (_Float16)acc[nb][j];
    }
  }
}

// ---------------------------------------------------------------------------
// Kernel 2: fused  q -> gather kk -> logits(WMMA) -> softmax -> out -> @wd_s -> sigmoid
// One wave per 16-row tile, 2 waves per block (58KB LDS).
// ---------------------------------------------------------------------------
__global__ void __launch_bounds__(64) attn_kernel(const float* __restrict__ data,
                                                  const int*   __restrict__ nidx,
                                                  const float* __restrict__ wq,
                                                  const float* __restrict__ wd,
                                                  const _Float16* __restrict__ kp,
                                                  float* __restrict__ out) {
  __shared__ __align__(16) _Float16 wqT[DD * DD];      // 8KB
  __shared__ __align__(16) _Float16 wdT[DD * DD];      // 8KB
  __shared__ __align__(16) _Float16 Kg[2][128 * DD];   // 32KB gathered neighbor rows
  __shared__ __align__(16) _Float16 Qh[2][16 * DD];    // 4KB  q tile (f16)
  __shared__ __align__(16) _Float16 oh[2][16 * DD];    // 4KB  out tile (f16)
  __shared__ float lg[2][128];                         // logits (site*8+k)
  __shared__ float aw[2][128];                         // attention weights

  const int tid  = threadIdx.x;
  const int lane = tid & 31;
  const int w    = tid >> 5;

  // cooperative f32->f16 transpose of wq and wd_s into LDS
#pragma unroll
  for (int i = 0; i < 64; ++i) {
    int id = i * 64 + tid;
    int n = id >> 6, k = id & 63;
    wqT[id] = (_Float16)wq[k * DD + n];
    wdT[id] = (_Float16)wd[k * DD + n];
  }

  const int tile  = blockIdx.x * 2 + w;
  const int bt    = tile / RT_PER_BT;
  const int rt    = tile % RT_PER_BT;
  const int rbase = rt * 16;
  const size_t n0 = (size_t)bt * RR + rbase;
  const float* tb = data + n0 * DD;

  // resolve gather addresses and kick off the async global->LDS DMA of the
  // 128 projected neighbor rows (ASYNCcnt-tracked; overlaps with Q GEMM below)
#pragma unroll
  for (int i = 0; i < 4; ++i) {
    int p = i * 32 + lane;                  // p = site*8 + k, 128 pairs per tile
    int site = p >> 3, k = p & 7;
    int nr = nidx[(rbase + site) * KK + k];
    const _Float16* src = kp + ((size_t)bt * RR + (size_t)nr) * DD;
    async_row128_to_lds(src, &Kg[w][p * DD]);
  }

  __syncthreads();   // wqT/wdT visible

  // ---- Q = data_tile @ wq  (8 WMMAs), stash as f16 in LDS
  {
    v8f accQ[4] = {};
#pragma unroll
    for (int s = 0; s < 2; ++s) {
      v16h A = a_frag_from_global_f32(tb, lane, s);
      v16h Bf[4];
#pragma unroll
      for (int nb = 0; nb < 4; ++nb) Bf[nb] = b_frag_from_lds(wqT, lane, nb, s);
#pragma unroll
      for (int nb = 0; nb < 4; ++nb) accQ[nb] = wmma_f16(A, Bf[nb], accQ[nb]);
    }
#pragma unroll
    for (int nb = 0; nb < 4; ++nb) {
      int col = nb * 16 + (lane & 15);
#pragma unroll
      for (int j = 0; j < 8; ++j) {
        int m = j + ((lane >> 4) << 3);
        Qh[w][m * DD + col] = (_Float16)accQ[nb][j];
      }
    }
  }

  wait_async_done();   // Kg DMA complete (compiler doesn't track the asm ops)
  __syncthreads();

  // ---- logits = Q(16x64) @ Kg^T(64x128)  (16 WMMAs); Kg row-major == B layout
  {
    v8f accL[8] = {};
#pragma unroll
    for (int s = 0; s < 2; ++s) {
      v16h A = a_frag_from_lds_f16(&Qh[w][0], lane, s);
#pragma unroll
      for (int g = 0; g < 2; ++g) {
        v16h Bf[4];
#pragma unroll
        for (int nb = 0; nb < 4; ++nb)
          Bf[nb] = b_frag_from_lds(&Kg[w][0], lane, g * 4 + nb, s);
#pragma unroll
        for (int nb = 0; nb < 4; ++nb)
          accL[g * 4 + nb] = wmma_f16(A, Bf[nb], accL[g * 4 + nb]);
      }
    }
    // keep only block-diagonal entries: site m needs columns p = m*8 .. m*8+7
#pragma unroll
    for (int nb = 0; nb < 8; ++nb) {
      int p = nb * 16 + (lane & 15);
#pragma unroll
      for (int j = 0; j < 8; ++j) {
        int m = j + ((lane >> 4) << 3);
        if ((p >> 3) == m) lg[w][(m << 3) + (p & 7)] = accL[nb][j] * 0.125f;  // /sqrt(64)
      }
    }
  }
  __syncthreads();

  // ---- softmax over K=8, one site per lane (lanes 0-15)
  if (lane < 16) {
    float v[8]; float mx = -1e30f;
#pragma unroll
    for (int k = 0; k < 8; ++k) { v[k] = lg[w][(lane << 3) + k]; mx = fmaxf(mx, v[k]); }
    float ssum = 0.f;
#pragma unroll
    for (int k = 0; k < 8; ++k) { v[k] = __expf(v[k] - mx); ssum += v[k]; }
    float inv = 1.f / ssum;
#pragma unroll
    for (int k = 0; k < 8; ++k) aw[w][(lane << 3) + k] = v[k] * inv;
  }
  __syncthreads();

  // ---- out = q + attn @ kk  (f32 VALU on LDS-resident Kg; 2 lanes per site)
  {
    const int site = lane >> 1;
    const int e0   = (lane & 1) << 5;
    float a[8];
#pragma unroll
    for (int k = 0; k < 8; ++k) a[k] = aw[w][(site << 3) + k];
#pragma unroll
    for (int c = 0; c < 4; ++c) {
      int e = e0 + c * 8;
      v8h qv = *(const v8h*)&Qh[w][site * DD + e];
      float o[8];
#pragma unroll
      for (int i = 0; i < 8; ++i) o[i] = (float)qv[i];
#pragma unroll
      for (int k = 0; k < 8; ++k) {
        v8h kv = *(const v8h*)&Kg[w][(site * 8 + k) * DD + e];
#pragma unroll
        for (int i = 0; i < 8; ++i) o[i] += a[k] * (float)kv[i];
      }
      v8h ov;
#pragma unroll
      for (int i = 0; i < 8; ++i) ov[i] = (_Float16)o[i];
      *(v8h*)&oh[w][site * DD + e] = ov;
    }
  }
  __syncthreads();

  // ---- final: sigmoid(out_tile @ wd_s)  (8 WMMAs), coalesced f32 stores
  {
    v8f accS[4] = {};
#pragma unroll
    for (int s = 0; s < 2; ++s) {
      v16h A = a_frag_from_lds_f16(&oh[w][0], lane, s);
      v16h Bf[4];
#pragma unroll
      for (int nb = 0; nb < 4; ++nb) Bf[nb] = b_frag_from_lds(wdT, lane, nb, s);
#pragma unroll
      for (int nb = 0; nb < 4; ++nb) accS[nb] = wmma_f16(A, Bf[nb], accS[nb]);
    }
#pragma unroll
    for (int nb = 0; nb < 4; ++nb) {
      int col = nb * 16 + (lane & 15);
#pragma unroll
      for (int j = 0; j < 8; ++j) {
        int m = j + ((lane >> 4) << 3);
        float x = accS[nb][j];
        out[(n0 + m) * DD + col] = 1.f / (1.f + __expf(-x));
      }
    }
  }
}

// ---------------------------------------------------------------------------
extern "C" void kernel_launch(void* const* d_in, const int* in_sizes, int n_in,
                              void* d_out, int out_size, void* d_ws, size_t ws_size,
                              hipStream_t stream) {
  const float* data = (const float*)d_in[0];   // (B,T,R,D) f32
  const int*   nidx = (const int*)  d_in[1];   // (R,K) int32
  const float* wq   = (const float*)d_in[2];   // (D,D) f32
  const float* wk   = (const float*)d_in[3];   // (D,D) f32
  const float* wd   = (const float*)d_in[4];   // (D,D) f32
  float* out = (float*)d_out;                  // (B,T,R,D) f32
  _Float16* kp = (_Float16*)d_ws;              // 240000*64 f16 = 30.72 MB scratch

  kproj_kernel<<<dim3(NTILES / 8), dim3(256), 0, stream>>>(data, wk, kp);
  attn_kernel <<<dim3(NTILES / 2), dim3(64),  0, stream>>>(data, nidx, wq, wd, kp, out);
}
